// APNB_17016660427253
// MI455X (gfx1250) — matrix operand order repaired and checked by reading, not compiled, not run on hardware
//
#include <hip/hip_runtime.h>

typedef __bf16 bf16_t;
typedef bf16_t v16bf __attribute__((ext_vector_type(16)));
typedef bf16_t v8bf  __attribute__((ext_vector_type(8)));
typedef bf16_t v2bf  __attribute__((ext_vector_type(2)));
typedef float  v8f   __attribute__((ext_vector_type(8)));

static __device__ __forceinline__ v8f wmma_bf16(v16bf a, v16bf b, v8f c) {
  return __builtin_amdgcn_wmma_f32_16x16x32_bf16(false, a, false, b, (short)0, c, false, false);
}

union AFrag { v16bf v; v8bf h[2]; };

#define B_   8
#define C_   512
#define HW_  96
#define N_   9216
#define O_   256
#define S_   110
#define SP_  128
#define MT_  128

// LDS strides (elements), chosen for 16B alignment + conflict-free b128 reads
#define SA   40    // x-chunk A tile [128][32] bf16 (double-buffered)
#define ABUF (MT_ * SA)
#define SQ   264   // q tile [128][256] bf16; reused as f32 energy [128][132]
#define SE   132
#define SATT 136   // attn [128][128pad] bf16
#define SOST 133   // out staging [32][128] f32 (also softmax reduction scratch)

// ---------------------------------------------------------------------------
// Kernel 1: PSP pooling  x[B,C,96,96] -> xp[B,C,110]
// bins: [0]=1x1, [1..9]=3x3, [10..45]=6x6, [46..109]=8x8
// ---------------------------------------------------------------------------
__global__ void psp_pool_kernel(const float* __restrict__ x, float* __restrict__ xp) {
  __shared__ float bins[110];
  const int t  = threadIdx.x;
  const int bc = blockIdx.x;
  if (t < 110) bins[t] = 0.0f;
  __syncthreads();
  const float* p = x + (size_t)bc * N_;
  const int start = t * 36;                 // 256 threads * 36 px = 9216
  int h = start / 96;
  int w = start - h * 96;
  int b8 = (h / 12) * 8 + w / 12;
  int b6 = (h / 16) * 6 + w / 16;
  int b3 = (h / 32) * 3 + w / 32;
  float a8 = 0.f, a6 = 0.f, a3 = 0.f, a1 = 0.f;
  for (int i = 0; i < 36; ++i) {
    const float v = p[start + i];
    a8 += v; a6 += v; a3 += v; a1 += v;
    ++w; if (w == 96) { w = 0; ++h; }
    if (i < 35) {  // run-length flush: few atomics, no hot-bin serialization
      const int n8 = (h / 12) * 8 + w / 12;
      const int n6 = (h / 16) * 6 + w / 16;
      const int n3 = (h / 32) * 3 + w / 32;
      if (n8 != b8) { atomicAdd(&bins[46 + b8], a8); a8 = 0.f; b8 = n8; }
      if (n6 != b6) { atomicAdd(&bins[10 + b6], a6); a6 = 0.f; b6 = n6; }
      if (n3 != b3) { atomicAdd(&bins[1  + b3], a3); a3 = 0.f; b3 = n3; }
    }
  }
  atomicAdd(&bins[46 + b8], a8);
  atomicAdd(&bins[10 + b6], a6);
  atomicAdd(&bins[1  + b3], a3);
  atomicAdd(&bins[0],       a1);
  __syncthreads();
  if (t < 110) {
    const float cnt = (t == 0) ? 9216.f : (t < 10) ? 1024.f : (t < 46) ? 256.f : 144.f;
    xp[(size_t)bc * S_ + t] = bins[t] * (1.0f / cnt);
  }
}

// ---------------------------------------------------------------------------
// Kernel 2: Wq fp32 -> bf16 (layout [O][C], K-contiguous per WMMA B column)
// ---------------------------------------------------------------------------
__global__ void wq_bf16_kernel(const float* __restrict__ Wq, bf16_t* __restrict__ Wqb) {
  const int i = blockIdx.x * 256 + threadIdx.x;
  if (i < O_ * C_) Wqb[i] = (bf16_t)Wq[i];
}

// ---------------------------------------------------------------------------
// Kernel 3: pooled K/V GEMMs (tiny; pooling commutes with conv1x1)
//   kb[b][s][o] = bf16(Wk[o,:]·xp[b,:,s] + bk[o])   (s>=110 -> 0)
//   vb[b][c][s] = bf16(Wv[c,:]·xp[b,:,s] + bv[c])   (s>=110 -> 0)
// ---------------------------------------------------------------------------
__global__ void kv_pool_gemm_kernel(const float* __restrict__ xp,
                                    const float* __restrict__ Wk, const float* __restrict__ bk,
                                    const float* __restrict__ Wv, const float* __restrict__ bv,
                                    bf16_t* __restrict__ kb, bf16_t* __restrict__ vb) {
  const int idx = blockIdx.x * 256 + threadIdx.x;
  const int NK = B_ * SP_ * O_;                       // 262144
  if (idx < NK) {
    const int o = idx & (O_ - 1);
    const int s = (idx >> 8) & (SP_ - 1);
    const int b = idx >> 15;
    float acc = 0.f;
    if (s < S_) {
      const float* w  = Wk + (size_t)o * C_;
      const float* xc = xp + (size_t)b * C_ * S_ + s;
      acc = bk[o];
      #pragma unroll 8
      for (int c = 0; c < C_; ++c) acc += w[c] * xc[(size_t)c * S_];
    }
    kb[((size_t)b * SP_ + s) * O_ + o] = (bf16_t)acc;
  } else {
    const int j = idx - NK;                           // < 524288
    const int s = j & (SP_ - 1);
    const int c = (j >> 7) & (C_ - 1);
    const int b = j >> 16;
    float acc = 0.f;
    if (s < S_) {
      const float* w  = Wv + (size_t)c * C_;
      const float* xc = xp + (size_t)b * C_ * S_ + s;
      acc = bv[c];
      #pragma unroll 8
      for (int cc = 0; cc < C_; ++cc) acc += w[cc] * xc[(size_t)cc * S_];
    }
    vb[((size_t)b * C_ + c) * SP_ + s] = (bf16_t)acc;
  }
}

// ---------------------------------------------------------------------------
// Kernel 4: fused attention per (b, 128-position tile). 256 threads = 8 waves.
// ---------------------------------------------------------------------------
__global__ __launch_bounds__(256) void apnb_attn_kernel(
    const float* __restrict__ x, const bf16_t* __restrict__ Wqb, const float* __restrict__ bq,
    const bf16_t* __restrict__ kb, const bf16_t* __restrict__ vb, float* __restrict__ out) {

  __shared__ __align__(16) bf16_t Als[2 * ABUF];     // 20.0 KB  x chunk, double buffered
  __shared__ __align__(16) bf16_t Qls[MT_ * SQ];     // 66.0 KB  q bf16 / energy f32
  __shared__ __align__(16) bf16_t ATT[MT_ * SATT];   // 34.0 KB  attn bf16
  __shared__ __align__(16) float  OST[32 * SOST];    // 16.6 KB  out staging / smax scratch
  __shared__ float BQ[O_];

  const int tid  = threadIdx.x;
  const int wave = tid >> 5;
  const int lane = tid & 31;
  const int lm   = lane & 15;   // row/col within fragment
  const int lg   = lane >> 4;   // half-wave K-group

  const int b  = blockIdx.x / (N_ / MT_);
  const int n0 = (blockIdx.x % (N_ / MT_)) * MT_;

  BQ[tid] = bq[tid];

  // ============ Phase A: q[128,256] = bf16(x_tile^T @ Wq^T + bq) -> Qls ====
  // software pipeline: global-load chunk k+1 -> WMMA chunk k -> cvt+store k+1
  const int mbA = (wave & 3) * 32;     // 2 M-tiles per wave
  const int obA = (wave >> 2) * 128;   // 8 N-tiles per wave
  v8f qacc[2][8];
  #pragma unroll
  for (int i = 0; i < 2; ++i)
    #pragma unroll
    for (int j = 0; j < 8; ++j) qacc[i][j] = (v8f)0.0f;

  // staging map: each thread owns a channel PAIR and 8 positions -> b32 stores
  const int cp  = (tid & 15) * 2;      // channel pair base 0..30
  const int mm8 = (tid >> 4) * 8;      // position base 0..120
  const float* xbase = x + (size_t)b * C_ * N_ + n0 + mm8;

  float fa[8], fb[8];
  {abort:;}
  // prologue: load + stage chunk 0
  {
    const float* x0 = xbase + (size_t)cp * N_;
    const float4 r0 = *(const float4*)(x0);
    const float4 r1 = *(const float4*)(x0 + 4);
    const float4 r2 = *(const float4*)(x0 + N_);
    const float4 r3 = *(const float4*)(x0 + N_ + 4);
    fa[0]=r0.x; fa[1]=r0.y; fa[2]=r0.z; fa[3]=r0.w; fa[4]=r1.x; fa[5]=r1.y; fa[6]=r1.z; fa[7]=r1.w;
    fb[0]=r2.x; fb[1]=r2.y; fb[2]=r2.z; fb[3]=r2.w; fb[4]=r3.x; fb[5]=r3.y; fb[6]=r3.z; fb[7]=r3.w;
    #pragma unroll
    for (int r = 0; r < 8; ++r) {
      v2bf p; p.x = (bf16_t)fa[r]; p.y = (bf16_t)fb[r];
      *(v2bf*)&Als[(mm8 + r) * SA + cp] = p;
    }
  }
  __syncthreads();

  for (int kc = 0; kc < C_ / 32; ++kc) {
    // (1) issue global loads for next chunk (latency hidden by WMMAs below)
    if (kc < C_ / 32 - 1) {
      const float* x1 = xbase + (size_t)((kc + 1) * 32 + cp) * N_;
      const float4 r0 = *(const float4*)(x1);
      const float4 r1 = *(const float4*)(x1 + 4);
      const float4 r2 = *(const float4*)(x1 + N_);
      const float4 r3 = *(const float4*)(x1 + N_ + 4);
      fa[0]=r0.x; fa[1]=r0.y; fa[2]=r0.z; fa[3]=r0.w; fa[4]=r1.x; fa[5]=r1.y; fa[6]=r1.z; fa[7]=r1.w;
      fb[0]=r2.x; fb[1]=r2.y; fb[2]=r2.z; fb[3]=r2.w; fb[4]=r3.x; fb[5]=r3.y; fb[6]=r3.z; fb[7]=r3.w;
    }
    // (2) WMMAs for current chunk
    const bf16_t* cur = Als + (kc & 1) * ABUF;
    AFrag a[2];
    #pragma unroll
    for (int i = 0; i < 2; ++i) {
      const int mr = mbA + i * 16 + lm;
      a[i].h[0] = *(const v8bf*)&cur[mr * SA + lg * 16];       // K = g*16..+7
      a[i].h[1] = *(const v8bf*)&cur[mr * SA + lg * 16 + 8];   // K = g*16+8..+15
    }
    #pragma unroll
    for (int j = 0; j < 8; ++j) {
      const v16bf bf = *(const v16bf*)(Wqb + (size_t)(obA + j * 16 + lm) * C_ +
                                       kc * 32 + lg * 16);
      #pragma unroll
      for (int i = 0; i < 2; ++i) qacc[i][j] = wmma_bf16(a[i].v, bf, qacc[i][j]);
    }
    // (3) convert + store next chunk into the other buffer
    if (kc < C_ / 32 - 1) {
      bf16_t* nxt = Als + ((kc + 1) & 1) * ABUF;
      #pragma unroll
      for (int r = 0; r < 8; ++r) {
        v2bf p; p.x = (bf16_t)fa[r]; p.y = (bf16_t)fb[r];
        *(v2bf*)&nxt[(mm8 + r) * SA + cp] = p;
      }
      __syncthreads();
    }
  }
  // dump q (+bias) to LDS
  #pragma unroll
  for (int j = 0; j < 8; ++j) {
    const int o = obA + j * 16 + lm;
    const float bias = BQ[o];
    #pragma unroll
    for (int i = 0; i < 2; ++i) {
      const int mr = mbA + i * 16 + lg * 8;
      #pragma unroll
      for (int r = 0; r < 8; ++r)
        Qls[(mr + r) * SQ + o] = (bf16_t)(qacc[i][j][r] + bias);
    }
  }
  __syncthreads();

  // ============ Phase B: energy[128,128pad] = q @ k^T (K=256) ==============
  const int mbB = (wave & 3) * 32;
  const int sbB = (wave >> 2) * 64;
  const bf16_t* kbp = kb + (size_t)b * SP_ * O_;
  v8f eacc[2][4];
  #pragma unroll
  for (int i = 0; i < 2; ++i)
    #pragma unroll
    for (int j = 0; j < 4; ++j) eacc[i][j] = (v8f)0.0f;

  for (int kc = 0; kc < O_ / 32; ++kc) {
    AFrag a[2];
    #pragma unroll
    for (int i = 0; i < 2; ++i) {
      const int mr = mbB + i * 16 + lm;
      a[i].h[0] = *(const v8bf*)&Qls[mr * SQ + kc * 32 + lg * 16];
      a[i].h[1] = *(const v8bf*)&Qls[mr * SQ + kc * 32 + lg * 16 + 8];
    }
    #pragma unroll
    for (int j = 0; j < 4; ++j) {
      const v16bf bf = *(const v16bf*)(kbp + (size_t)(sbB + j * 16 + lm) * O_ +
                                       kc * 32 + lg * 16);
      #pragma unroll
      for (int i = 0; i < 2; ++i) eacc[i][j] = wmma_bf16(a[i].v, bf, eacc[i][j]);
    }
  }
  __syncthreads();                       // all Qls reads complete
  float* Els = (float*)Qls;              // reuse as f32 [128][132]
  #pragma unroll
  for (int j = 0; j < 4; ++j) {
    const int s = sbB + j * 16 + lm;
    #pragma unroll
    for (int i = 0; i < 2; ++i) {
      const int mr = mbB + i * 16 + lg * 8;
      #pragma unroll
      for (int r = 0; r < 8; ++r) Els[(mr + r) * SE + s] = eacc[i][j][r];
    }
  }
  __syncthreads();

  // ============ softmax over s<110 (2 threads/row), bf16 attn, zero pad ====
  {
    float* RED = OST;                    // reuse out-staging as reduction scratch
    const int row  = tid & 127;
    const int half = tid >> 7;
    const int sB   = half * 55;
    const int sE   = sB + 55;            // 110 = 55 + 55
    float* er = Els + row * SE;
    float mx = -3.0e38f;
    for (int s = sB; s < sE; ++s) mx = fmaxf(mx, er[s]);
    RED[tid] = mx;
    __syncthreads();
    mx = fmaxf(RED[row], RED[row + 128]);
    float sum = 0.f;
    for (int s = sB; s < sE; ++s) { const float e = __expf(er[s] - mx); er[s] = e; sum += e; }
    __syncthreads();                     // max reads done before overwrite
    RED[tid] = sum;
    __syncthreads();
    const float inv = 1.0f / (RED[row] + RED[row + 128]);
    bf16_t* ar = ATT + row * SATT;
    for (int s = sB; s < sE; ++s) ar[s] = (bf16_t)(er[s] * inv);
    if (half) for (int s = S_; s < SATT; ++s) ar[s] = (bf16_t)0.0f;
  }
  __syncthreads();

  // ============ Phase C: out[128,512] = attn @ v^T + x (32-ch chunks) ======
  const int mbC = (wave & 3) * 32;
  const int cbC = (wave >> 2) * 16;
  const bf16_t* vbp = vb + (size_t)b * C_ * SP_;
  const int c2  = tid >> 3;            // store: channel-in-chunk 0..31
  const int mm  = (tid & 7) * 16;      // store: position base

  for (int cc = 0; cc < C_ / 32; ++cc) {
    v8f oacc[2];
    oacc[0] = (v8f)0.0f; oacc[1] = (v8f)0.0f;
    #pragma unroll
    for (int kc = 0; kc < SP_ / 32; ++kc) {
      AFrag a[2];
      #pragma unroll
      for (int i = 0; i < 2; ++i) {
        const int mr = mbC + i * 16 + lm;
        a[i].h[0] = *(const v8bf*)&ATT[mr * SATT + kc * 32 + lg * 16];
        a[i].h[1] = *(const v8bf*)&ATT[mr * SATT + kc * 32 + lg * 16 + 8];
      }
      const v16bf bf = *(const v16bf*)(vbp + (size_t)(cc * 32 + cbC + lm) * SP_ +
                                       kc * 32 + lg * 16);
      #pragma unroll
      for (int i = 0; i < 2; ++i) oacc[i] = wmma_bf16(a[i].v, bf, oacc[i]);
    }
    __syncthreads();                   // previous OST reads complete
    const int clo = cbC + lm;
    #pragma unroll
    for (int i = 0; i < 2; ++i) {
      const int mr = mbC + i * 16 + lg * 8;
      #pragma unroll
      for (int r = 0; r < 8; ++r) OST[clo * SOST + mr + r] = oacc[i][r];
    }
    __syncthreads();
    // coalesced store with residual add
    const size_t gb = ((size_t)(b * C_ + cc * 32 + c2)) * N_ + n0 + mm;
    #pragma unroll
    for (int v = 0; v < 4; ++v) {
      const float4 xr = *(const float4*)(x + gb + v * 4);
      float4 ov;
      ov.x = OST[c2 * SOST + mm + v * 4 + 0] + xr.x;
      ov.y = OST[c2 * SOST + mm + v * 4 + 1] + xr.y;
      ov.z = OST[c2 * SOST + mm + v * 4 + 2] + xr.z;
      ov.w = OST[c2 * SOST + mm + v * 4 + 3] + xr.w;
      *(float4*)(out + gb + v * 4) = ov;
    }
  }
}

// ---------------------------------------------------------------------------
extern "C" void kernel_launch(void* const* d_in, const int* in_sizes, int n_in,
                              void* d_out, int out_size, void* d_ws, size_t ws_size,
                              hipStream_t stream) {
  const float* x  = (const float*)d_in[0];
  const float* Wq = (const float*)d_in[1];
  const float* bq = (const float*)d_in[2];
  const float* Wk = (const float*)d_in[3];
  const float* bk = (const float*)d_in[4];
  const float* Wv = (const float*)d_in[5];
  const float* bv = (const float*)d_in[6];
  float* out = (float*)d_out;

  char* ws = (char*)d_ws;
  const size_t XP_BYTES  = (size_t)B_ * C_ * S_ * 4;   // 1,802,240
  const size_t WQB_BYTES = (size_t)O_ * C_ * 2;        //   262,144
  const size_t KB_BYTES  = (size_t)B_ * SP_ * O_ * 2;  //   524,288
  float*  xp  = (float*)ws;
  bf16_t* Wqb = (bf16_t*)(ws + XP_BYTES);
  bf16_t* kbp = (bf16_t*)(ws + XP_BYTES + WQB_BYTES);
  bf16_t* vbp = (bf16_t*)(ws + XP_BYTES + WQB_BYTES + KB_BYTES);

  psp_pool_kernel<<<B_ * C_, 256, 0, stream>>>(x, xp);
  wq_bf16_kernel<<<(O_ * C_) / 256, 256, 0, stream>>>(Wq, Wqb);
  kv_pool_gemm_kernel<<<(B_ * SP_ * O_ + B_ * C_ * SP_) / 256, 256, 0, stream>>>(
      xp, Wk, bk, Wv, bv, kbp, vbp);
  apnb_attn_kernel<<<B_ * (N_ / MT_), 256, 0, stream>>>(x, Wqb, bq, kbp, vbp, out);
}
